// HRAN_ConvD_14164802142394
// MI455X (gfx1250) — matrix operand order repaired and checked by reading, not compiled, not run on hardware
//
#include <hip/hip_runtime.h>
#include <hip/hip_bf16.h>

typedef __attribute__((ext_vector_type(16))) __bf16 v16bf;
typedef __attribute__((ext_vector_type(8)))  float  v8f;

#define B_      1024
#define NENT    40000
#define KDIM    200
#define NFILT   200
#define NPOS    194
#define FLAT    38800      // NFILT * NPOS
#define NCHUNK  1213       // ceil(FLAT / 32)
#define KSPLIT  8
#define CPB     152        // ceil(NCHUNK / KSPLIT)
#define NPAD    208        // 13 * 16  (partial buffer cols: valid FC outputs)
#define NB2     256        // 16 * 16  (B tile rows incl. zero pad -> branch-free loop)
#define KYPAD   224        // 7 * 32   (y width padded for GEMM2 K loop)
#define EPS_    1e-5f

struct U32x8 { uint4 a, b; };

__device__ __forceinline__ v16bf load_frag_b128x2(const __bf16* p0, const __bf16* p1) {
    U32x8 u;
    u.a = *(const uint4*)p0;
    u.b = *(const uint4*)p1;
    return __builtin_bit_cast(v16bf, u);
}

// -------------------------------------------------------------------------
// Kernel 1: fused  gather -> BN0 -> conv -> BN1 -> ReLU -> (bf16) -> WMMA FC
// Grid: (64 M-tiles, 8 K-splits), 128 threads (4 waves).
// N padded to 256 (zero B rows for n>=200) so each wave runs 4 unconditional
// WMMAs per K-chunk: branch-free inner loop, no acc phi copies, EXEC all-1s.
// Writes f32 partial sums part[ks][1024][208] (deterministic reduction later).
// -------------------------------------------------------------------------
__global__ __launch_bounds__(128)
void k_feat_gemm1(const int* __restrict__ sub, const int* __restrict__ rel,
                  const float* __restrict__ emb, const float* __restrict__ remb,
                  const float* __restrict__ conv_w, const float* __restrict__ conv_b,
                  const float* __restrict__ g0, const float* __restrict__ b0,
                  const float* __restrict__ mm0, const float* __restrict__ vv0,
                  const float* __restrict__ g1, const float* __restrict__ b1,
                  const float* __restrict__ mm1, const float* __restrict__ vv1,
                  const float* __restrict__ fcw, float* __restrict__ part)
{
    __shared__ float  hlds[16 * 2 * 208];     // BN0-normalized inputs per M row
    __shared__ float  wlds[NFILT * 14];       // conv weights [f][c][j]
    __shared__ float  s1l[NFILT], t1l[NFILT]; // folded conv_b + BN1
    __shared__ __bf16 Al[16 * 40];            // A tile 16 x 32 (stride 40)
    __shared__ __bf16 Bl[NB2 * 40];           // B tile transposed: [n][k]
    __shared__ int    ids[16], idr[16];

    const int tid   = threadIdx.x;
    const int mbase = blockIdx.x * 16;
    const int ks    = blockIdx.y;

    if (tid < 16) { ids[tid] = sub[mbase + tid]; idr[tid] = rel[mbase + tid]; }
    __syncthreads();

    const float s00 = g0[0] * rsqrtf(vv0[0] + EPS_);
    const float t00 = b0[0] - mm0[0] * s00;
    const float s01 = g0[1] * rsqrtf(vv0[1] + EPS_);
    const float t01 = b0[1] - mm0[1] * s01;

    for (int idx = tid; idx < 16 * 2 * 208; idx += 128) {
        int m = idx / 416, r = idx - m * 416;
        int c = r / 208,   k = r - c * 208;
        float v = 0.f;
        if (k < KDIM) {
            int row = c ? idr[m] : ids[m];
            float e = c ? remb[row * KDIM + k] : emb[row * KDIM + k];
            v = c ? (e * s01 + t01) : (e * s00 + t00);
        }
        hlds[idx] = v;
    }
    for (int idx = tid; idx < NFILT * 14; idx += 128) wlds[idx] = conv_w[idx];
    for (int f = tid; f < NFILT; f += 128) {
        float s = g1[f] * rsqrtf(vv1[f] + EPS_);
        s1l[f] = s;
        t1l[f] = (conv_b[f] - mm1[f]) * s + b1[f];  // conv bias folded into BN1
    }
    __syncthreads();

    const int lane  = tid & 31;
    const int uwave = __builtin_amdgcn_readfirstlane(tid >> 5);  // SGPR wave id
    const int half  = lane >> 4, lm = lane & 15;

    v8f acc[4] = {};

    const int ck0 = ks * CPB;
    const int ck1 = (ck0 + CPB < NCHUNK) ? ck0 + CPB : NCHUNK;

    for (int ck = ck0; ck < ck1; ++ck) {
        const int kbase = ck * 32;

        // ---- A tile: conv + BN1 + ReLU computed on the fly (4 elems/thread)
        #pragma unroll
        for (int i = 0; i < 4; ++i) {
            int e = tid * 4 + i;
            int m = e >> 5, kl = e & 31;
            unsigned kk = (unsigned)(kbase + kl);
            float v = 0.f;
            if (kk < FLAT) {
                unsigned f = kk / 194u;
                unsigned p = kk - f * 194u;
                const float* hr = &hlds[m * 416 + p];
                const float* wr = &wlds[f * 14];
                float a = 0.f;
                #pragma unroll
                for (int j = 0; j < 7; ++j) a += hr[j] * wr[j];
                #pragma unroll
                for (int j = 0; j < 7; ++j) a += hr[208 + j] * wr[7 + j];
                v = fmaxf(a * s1l[f] + t1l[f], 0.f);
            }
            Al[m * 40 + kl] = (__bf16)v;
        }
        // ---- B tile: fc_w[kk][n] -> Bl[n][kl] (transpose, f32 -> bf16)
        // 256 rows, zeros for n >= 200; index math is pure shifts.
        for (int idx = tid; idx < NB2 * 32; idx += 128) {
            int kl = idx >> 8, n = idx & 255;
            unsigned kk = (unsigned)(kbase + kl);
            float v = (kk < FLAT && n < KDIM) ? fcw[kk * KDIM + n] : 0.f;
            Bl[n * 40 + kl] = (__bf16)v;
        }
        __syncthreads();

        const __bf16* ar = &Al[lm * 40];
        v16bf afrag = load_frag_b128x2(ar + half * 8, ar + 16 + half * 8);
        v16bf bfrag[4];
        #pragma unroll
        for (int q = 0; q < 4; ++q) {
            const __bf16* br = &Bl[((uwave * 4 + q) * 16 + lm) * 40];
            bfrag[q] = load_frag_b128x2(br + half * 8, br + 16 + half * 8);
        }
        #pragma unroll
        for (int q = 0; q < 4; ++q) {
            acc[q] = __builtin_amdgcn_wmma_f32_16x16x32_bf16(
                false, afrag, false, bfrag[q], (short)0, acc[q], false, false);
        }
        __syncthreads();
    }

    float* pbase = part + (size_t)ks * B_ * NPAD;
    #pragma unroll
    for (int q = 0; q < 4; ++q) {
        int ns = uwave * 4 + q;
        if (ns < 13) {              // one-time scalar guard (outside hot loop)
            int n = ns * 16 + lm;
            #pragma unroll
            for (int r = 0; r < 8; ++r) {
                int mg = mbase + half * 8 + r;
                pbase[(size_t)mg * NPAD + n] = acc[q][r];
            }
        }
    }
}

// -------------------------------------------------------------------------
// Kernel 2: deterministic K-split reduction + fc_b + BN2 + ReLU -> bf16 y
// y stored padded to KYPAD=224 (zero tail) for GEMM2's 32-wide K loop.
// -------------------------------------------------------------------------
__global__ __launch_bounds__(256)
void k_reduce_bn2(const float* __restrict__ part,
                  const float* __restrict__ g2, const float* __restrict__ b2,
                  const float* __restrict__ mm2, const float* __restrict__ vv2,
                  const float* __restrict__ fcb, __bf16* __restrict__ yws)
{
    int idx = blockIdx.x * 256 + threadIdx.x;
    if (idx >= B_ * KYPAD) return;
    int m = idx / KYPAD, n = idx - m * KYPAD;
    float v = 0.f;
    if (n < KDIM) {
        float s = 0.f;
        #pragma unroll
        for (int ks = 0; ks < KSPLIT; ++ks)
            s += part[((size_t)ks * B_ + m) * NPAD + n];
        float sc = g2[n] * rsqrtf(vv2[n] + EPS_);
        v = fmaxf(s * sc + (fcb[n] - mm2[n]) * sc + b2[n], 0.f);
    }
    yws[idx] = (__bf16)v;
}

// -------------------------------------------------------------------------
// Kernel 3: scores = sigmoid(y @ embed^T).  M=1024, N=40000, K=224 (padded).
// Grid: (64 M-tiles, 79 N-tiles of 512), 256 threads (8 waves x 4 subtiles).
// Output write (164 MB) is the dominant memory stream; embed stays L2-resident.
// -------------------------------------------------------------------------
__global__ __launch_bounds__(256)
void k_scores(const __bf16* __restrict__ yws, const float* __restrict__ emb,
              float* __restrict__ out)
{
    __shared__ __bf16 Al[16 * 40];
    __shared__ __bf16 Bl[512 * 40];

    const int tid   = threadIdx.x;
    const int mbase = blockIdx.x * 16;
    const int nbase = blockIdx.y * 512;
    const int lane  = tid & 31;
    const int uwave = __builtin_amdgcn_readfirstlane(tid >> 5);  // SGPR wave id
    const int half  = lane >> 4, lm = lane & 15;

    v8f acc[4] = {};

    for (int kc = 0; kc < 7; ++kc) {
        const int kbase = kc * 32;
        for (int idx = tid; idx < 512; idx += 256) {
            int m = idx >> 5, kl = idx & 31;
            Al[m * 40 + kl] = yws[(mbase + m) * KYPAD + kbase + kl];
        }
        for (int idx = tid; idx < 512 * 32; idx += 256) {
            int nn = idx >> 5, kl = idx & 31;
            int ng = nbase + nn, kg = kbase + kl;
            float v = (ng < NENT && kg < KDIM) ? emb[(size_t)ng * KDIM + kg] : 0.f;
            Bl[nn * 40 + kl] = (__bf16)v;
        }
        __syncthreads();

        const __bf16* ar = &Al[lm * 40];
        v16bf afrag = load_frag_b128x2(ar + half * 8, ar + 16 + half * 8);
        v16bf bfrag[4];
        #pragma unroll
        for (int q = 0; q < 4; ++q) {
            const __bf16* br = &Bl[((uwave * 4 + q) * 16 + lm) * 40];
            bfrag[q] = load_frag_b128x2(br + half * 8, br + 16 + half * 8);
        }
        #pragma unroll
        for (int q = 0; q < 4; ++q) {
            acc[q] = __builtin_amdgcn_wmma_f32_16x16x32_bf16(
                false, afrag, false, bfrag[q], (short)0, acc[q], false, false);
        }
        __syncthreads();
    }

    #pragma unroll
    for (int q = 0; q < 4; ++q) {
        int n = nbase + (uwave * 4 + q) * 16 + lm;
        if (n < NENT) {
            #pragma unroll
            for (int r = 0; r < 8; ++r) {
                int mg = mbase + half * 8 + r;
                float x = acc[q][r];
                out[(size_t)mg * NENT + n] = 1.f / (1.f + __expf(-x));
            }
        }
    }
}

extern "C" void kernel_launch(void* const* d_in, const int* in_sizes, int n_in,
                              void* d_out, int out_size, void* d_ws, size_t ws_size,
                              hipStream_t stream)
{
    (void)in_sizes; (void)n_in; (void)out_size; (void)ws_size;
    const int*   sub  = (const int*)d_in[0];
    const int*   rel  = (const int*)d_in[1];
    const float* emb  = (const float*)d_in[2];
    const float* remb = (const float*)d_in[3];
    const float* cw   = (const float*)d_in[4];
    const float* cb   = (const float*)d_in[5];
    const float* g0 = (const float*)d_in[6],  *b0 = (const float*)d_in[7];
    const float* m0 = (const float*)d_in[8],  *v0 = (const float*)d_in[9];
    const float* g1 = (const float*)d_in[10], *b1 = (const float*)d_in[11];
    const float* m1 = (const float*)d_in[12], *v1 = (const float*)d_in[13];
    const float* g2 = (const float*)d_in[14], *b2 = (const float*)d_in[15];
    const float* m2 = (const float*)d_in[16], *v2 = (const float*)d_in[17];
    const float* fcw = (const float*)d_in[18], *fcb = (const float*)d_in[19];

    float*  part = (float*)d_ws;                                   // 8*1024*208 f32 (~6.8 MB)
    __bf16* yws  = (__bf16*)((char*)d_ws + (size_t)KSPLIT * B_ * NPAD * sizeof(float));

    k_feat_gemm1<<<dim3(64, KSPLIT), 128, 0, stream>>>(
        sub, rel, emb, remb, cw, cb, g0, b0, m0, v0, g1, b1, m1, v1, fcw, part);
    k_reduce_bn2<<<dim3((B_ * KYPAD) / 256), 256, 0, stream>>>(
        part, g2, b2, m2, v2, fcb, yws);
    k_scores<<<dim3(64, 79), 256, 0, stream>>>(yws, emb, (float*)d_out);
}
